// Baseline_73057393705690
// MI455X (gfx1250) — compile-verified, matrix operand order checked by reading, not run on hardware
//
#include <hip/hip_runtime.h>
#include <math.h>

#define N_LAYERS 24
#define D_MODEL  768
#define D_INNER  1536
#define D_STATE  16
#define DT_RANK  48
#define CONV_KW  4
#define VOCAB    50280
#define SEQ      8
#define BATCH    64
#define NTOK     (BATCH * SEQ)      // 512 tokens
#define FULL_L   64                 // full_ids / full_mask row length
#define XPROJ_N  (DT_RANK + 2 * D_STATE)   // 80

// ---------- CDNA5 WMMA types ----------
typedef __attribute__((ext_vector_type(16))) __bf16        v16bf;
typedef __attribute__((ext_vector_type(8)))  float         v8f;
typedef __attribute__((ext_vector_type(8)))  unsigned int  v8u;

__device__ __forceinline__ unsigned short f2bf_bits(float f) {
    union { float f; unsigned u; } c; c.f = f;
    unsigned u = c.u;
    unsigned r = u + 0x7FFFu + ((u >> 16) & 1u);   // round-to-nearest-even
    return (unsigned short)(r >> 16);
}
__device__ __forceinline__ unsigned pack2bf(float a, float b) {
    return (unsigned)f2bf_bits(a) | ((unsigned)f2bf_bits(b) << 16);
}

// =====================================================================
// GEMM: C[M,N] = epilog( A[M,K] * W[N,K]^T )
//   A row-major (lda), W row-major (ldb): both operands K-contiguous,
//   matching the CDNA5 16-bit A/B fragment layout exactly.
// Block: 256 threads = 8 waves arranged 4(M) x 2(N).
// Block tile 64(M) x 128(N); each wave owns a 16x64 slab:
//   1 A fragment reused across 4 B fragments -> 4 WMMAs per 32-K chunk.
// Fast path (block-uniform): unguarded float4 global loads + packed
// bf16 ds_store_b32.  Edge blocks / ragged K chunks take a guarded path.
// mode: 0 = plain store, 1 = C += acc (residual), 2 = softplus(acc+bias[n])
// Requires M % 64 == 0 (always true here: M = 512).
// =====================================================================
__global__ __launch_bounds__(256) void gemm_wmma_bf16(
    const float* __restrict__ A, const float* __restrict__ W,
    float* __restrict__ C, int M, int N, int K,
    int lda, int ldb, int ldc,
    const float* __restrict__ bias, int mode)
{
    __shared__ unsigned short As[64 * 34];    // stride 34: pair-aligned, bank-spread
    __shared__ unsigned short Bs[128 * 34];

    const int tid  = threadIdx.x;
    const int lane = tid & 31;
    const int wave = tid >> 5;
    const int wm   = wave >> 1;              // 0..3  (M sub-tile)
    const int wn   = wave & 1;               // 0..1  (N half)
    const int m0   = blockIdx.y * 64;
    const int n0   = blockIdx.x * 128;

    const int hi8  = (lane >= 16) ? 8 : 0;
    const int r    = lane & 15;

    const bool nFull = (n0 + 128 <= N);

    v8f acc[4] = {{}, {}, {}, {}};

    for (int k0 = 0; k0 < K; k0 += 32) {
        const bool kFull = (k0 + 32 <= K);
        if (nFull && kFull) {
            // ---------- fast path: float4 loads, packed bf16 stores ----------
            // A tile 64 x 32 : 512 float4, 2 per thread
            #pragma unroll
            for (int it = 0; it < 2; ++it) {
                int g   = tid + it * 256;
                int row = g >> 3, cg = (g & 7) * 4;
                float4 x = *reinterpret_cast<const float4*>(
                               &A[(size_t)(m0 + row) * lda + k0 + cg]);
                *(unsigned*)&As[row * 34 + cg]     = pack2bf(x.x, x.y);
                *(unsigned*)&As[row * 34 + cg + 2] = pack2bf(x.z, x.w);
            }
            // B tile 128 x 32 : 1024 float4, 4 per thread
            #pragma unroll
            for (int it = 0; it < 4; ++it) {
                int g   = tid + it * 256;
                int row = g >> 3, cg = (g & 7) * 4;
                float4 x = *reinterpret_cast<const float4*>(
                               &W[(size_t)(n0 + row) * ldb + k0 + cg]);
                *(unsigned*)&Bs[row * 34 + cg]     = pack2bf(x.x, x.y);
                *(unsigned*)&Bs[row * 34 + cg + 2] = pack2bf(x.z, x.w);
            }
        } else {
            // ---------- guarded path (edge N block or ragged K chunk) ----------
            #pragma unroll
            for (int it = 0; it < 8; ++it) {
                int idx = tid + it * 256;
                int row = idx >> 5, col = idx & 31;
                int gk = k0 + col;
                float x = (gk < K) ? A[(size_t)(m0 + row) * lda + gk] : 0.0f;
                As[row * 34 + col] = f2bf_bits(x);
            }
            #pragma unroll
            for (int it = 0; it < 16; ++it) {
                int idx = tid + it * 256;
                int row = idx >> 5, col = idx & 31;
                int gn = n0 + row, gk = k0 + col;
                float x = (gn < N && gk < K) ? W[(size_t)gn * ldb + gk] : 0.0f;
                Bs[row * 34 + col] = f2bf_bits(x);
            }
        }
        __syncthreads();

        // ---- fragments per ISA 16-bit layout:
        //      reg pair v holds K = 2v + (v>=4 ? 8:0) + (lane>=16 ? 8:0) ----
        union { v8u u; v16bf b; } fa, fb[4];
        const unsigned short* ar = &As[(wm * 16 + r) * 34];
        #pragma unroll
        for (int vv = 0; vv < 8; ++vv) {
            int kb = 2 * vv + ((vv >= 4) ? 8 : 0) + hi8;   // even -> dword aligned
            fa.u[vv] = *(const unsigned int*)(ar + kb);
        }
        #pragma unroll
        for (int nt = 0; nt < 4; ++nt) {
            const unsigned short* br = &Bs[(wn * 64 + nt * 16 + r) * 34];
            #pragma unroll
            for (int vv = 0; vv < 8; ++vv) {
                int kb = 2 * vv + ((vv >= 4) ? 8 : 0) + hi8;
                fb[nt].u[vv] = *(const unsigned int*)(br + kb);
            }
        }
        #pragma unroll
        for (int nt = 0; nt < 4; ++nt) {
            acc[nt] = __builtin_amdgcn_wmma_f32_16x16x32_bf16(
                          false, fa.b, false, fb[nt].b, (short)0, acc[nt],
                          false, false);
        }
        __syncthreads();
    }

    // ---- writeback: C/D layout -> M = i + 8*(lane>=16), N = lane&15 ----
    const int cmBase = m0 + wm * 16 + ((lane >> 4) << 3);
    const int cn     = lane & 15;

    #pragma unroll
    for (int nt = 0; nt < 4; ++nt) {
        int gn = n0 + wn * 64 + nt * 16 + cn;
        if (gn >= N) continue;
        float bval = (mode == 2) ? bias[gn] : 0.0f;
        #pragma unroll
        for (int i = 0; i < 8; ++i) {
            int m = cmBase + i;
            float x = acc[nt][i];
            if (mode == 2) {
                float t = x + bval;
                x = (t > 20.0f) ? t : logf(1.0f + __expf(t));   // softplus
            }
            float* p = &C[(size_t)m * ldc + gn];
            if (mode == 1) x += *p;
            *p = x;
        }
    }
}

// =====================================================================
// Embedding gather: h[bt, d] = embed[ids[b, t], d]
// =====================================================================
__global__ void embed_kernel(const int* __restrict__ ids,
                             const float* __restrict__ emb,
                             float* __restrict__ h)
{
    int idx = blockIdx.x * blockDim.x + threadIdx.x;
    if (idx >= NTOK * D_MODEL) return;
    int bt = idx / D_MODEL, d = idx % D_MODEL;
    int b = bt / SEQ, t = bt % SEQ;
    int id = ids[b * FULL_L + t];
    h[idx] = emb[(size_t)id * D_MODEL + d];
}

// =====================================================================
// RMSNorm: one block (256 thr) per token row
// =====================================================================
__global__ __launch_bounds__(256) void rmsnorm_kernel(
    const float* __restrict__ x, const float* __restrict__ w,
    float* __restrict__ out, int D)
{
    __shared__ float red[256];
    int row = blockIdx.x;
    const float* xr = x + (size_t)row * D;
    float s = 0.0f;
    for (int i = threadIdx.x; i < D; i += 256) { float v = xr[i]; s += v * v; }
    red[threadIdx.x] = s;
    __syncthreads();
    for (int off = 128; off > 0; off >>= 1) {
        if (threadIdx.x < off) red[threadIdx.x] += red[threadIdx.x + off];
        __syncthreads();
    }
    float rs = rsqrtf(red[0] / (float)D + 1e-5f);
    for (int i = threadIdx.x; i < D; i += 256)
        out[(size_t)row * D + i] = xr[i] * rs * w[i];
}

// =====================================================================
// Depthwise causal conv(K=4) + bias + SiLU + mask.
// x is the first D_INNER columns of xz[NTOK, 2*D_INNER]; x pre-masked.
// =====================================================================
__global__ void conv_silu_kernel(const float* __restrict__ xz,
                                 const int* __restrict__ mask,
                                 const float* __restrict__ cw,  // [D_INNER, 4]
                                 const float* __restrict__ cb,  // [D_INNER]
                                 float* __restrict__ xc)        // [NTOK, D_INNER]
{
    int idx = blockIdx.x * blockDim.x + threadIdx.x;
    if (idx >= NTOK * D_INNER) return;
    int bt = idx / D_INNER, d = idx % D_INNER;
    int b = bt / SEQ, t = bt % SEQ;
    float s = 0.0f;
    #pragma unroll
    for (int k = 0; k < CONV_KW; ++k) {
        int tt = t - (CONV_KW - 1) + k;
        if (tt >= 0) {
            float mv = (float)mask[b * FULL_L + tt];
            s += cw[d * CONV_KW + k] *
                 xz[(size_t)(b * SEQ + tt) * (2 * D_INNER) + d] * mv;
        }
    }
    s += cb[d];
    float sl = s / (1.0f + __expf(-s));              // SiLU
    xc[idx] = sl * (float)mask[b * FULL_L + t];
}

// =====================================================================
// Selective scan: one thread per (b, d); 16-state recurrence over T=8,
// fused with D-skip connection and SiLU(z) gating.
// =====================================================================
__global__ void scan_kernel(const float* __restrict__ dt,    // [NTOK, D_INNER]
                            const float* __restrict__ xc,    // [NTOK, D_INNER]
                            const float* __restrict__ xz,    // [NTOK, 2*D_INNER] (z at +D_INNER)
                            const float* __restrict__ proj,  // [NTOK, 80]  (B at 48, C at 64)
                            const float* __restrict__ A_log, // [D_INNER, 16]
                            const float* __restrict__ Dp,    // [D_INNER]
                            float* __restrict__ y)           // [NTOK, D_INNER]
{
    int idx = blockIdx.x * blockDim.x + threadIdx.x;
    if (idx >= BATCH * D_INNER) return;
    int b = idx / D_INNER, d = idx % D_INNER;

    float Arow[D_STATE];
    #pragma unroll
    for (int n = 0; n < D_STATE; ++n)
        Arow[n] = -__expf(A_log[(size_t)d * D_STATE + n]);
    float Dv = Dp[d];

    float st[D_STATE];
    #pragma unroll
    for (int n = 0; n < D_STATE; ++n) st[n] = 0.0f;

    for (int t = 0; t < SEQ; ++t) {
        int bt = b * SEQ + t;
        float dtv = dt[(size_t)bt * D_INNER + d];
        float xv  = xc[(size_t)bt * D_INNER + d];
        float zv  = xz[(size_t)bt * (2 * D_INNER) + D_INNER + d];
        const float* Bp = &proj[(size_t)bt * XPROJ_N + DT_RANK];
        const float* Cp = Bp + D_STATE;
        float yv = 0.0f;
        #pragma unroll
        for (int n = 0; n < D_STATE; ++n) {
            float dA = __expf(dtv * Arow[n]);
            st[n] = dA * st[n] + dtv * Bp[n] * xv;
            yv += st[n] * Cp[n];
        }
        yv += xv * Dv;
        float sz = zv / (1.0f + __expf(-zv));        // SiLU(z)
        y[(size_t)bt * D_INNER + d] = yv * sz;
    }
}

// =====================================================================
// Host driver
// =====================================================================
extern "C" void kernel_launch(void* const* d_in, const int* in_sizes, int n_in,
                              void* d_out, int out_size, void* d_ws, size_t ws_size,
                              hipStream_t stream)
{
    (void)in_sizes; (void)n_in; (void)out_size; (void)ws_size;

    const int*   ids    = (const int*)  d_in[0];
    const int*   mask   = (const int*)  d_in[1];
    /* d_in[2] full_loss_mask: unused by reference forward */
    const float* embedW = (const float*)d_in[3];
    const float* norm_w = (const float*)d_in[4];
    const float* in_w   = (const float*)d_in[5];
    const float* conv_w = (const float*)d_in[6];
    const float* conv_b = (const float*)d_in[7];
    const float* xp_w   = (const float*)d_in[8];
    const float* dtp_w  = (const float*)d_in[9];
    const float* dtp_b  = (const float*)d_in[10];
    const float* A_log  = (const float*)d_in[11];
    const float* Dp     = (const float*)d_in[12];
    const float* out_w  = (const float*)d_in[13];
    const float* normf  = (const float*)d_in[14];
    float* logits = (float*)d_out;

    // workspace carve-out (~19 MB)
    char*  ws  = (char*)d_ws;
    size_t off = 0;
    auto carve = [&](size_t nfloats) -> float* {
        float* p = (float*)(ws + off);
        off += ((nfloats * sizeof(float) + 255) & ~(size_t)255);
        return p;
    };
    float* h   = carve((size_t)NTOK * D_MODEL);
    float* hn  = carve((size_t)NTOK * D_MODEL);
    float* xzb = carve((size_t)NTOK * 2 * D_INNER);
    float* xcb = carve((size_t)NTOK * D_INNER);
    float* prj = carve((size_t)NTOK * XPROJ_N);
    float* dtb = carve((size_t)NTOK * D_INNER);
    float* yb  = carve((size_t)NTOK * D_INNER);

    embed_kernel<<<(NTOK * D_MODEL + 255) / 256, 256, 0, stream>>>(ids, embedW, h);

    for (int l = 0; l < N_LAYERS; ++l) {
        rmsnorm_kernel<<<NTOK, 256, 0, stream>>>(h, norm_w + (size_t)l * D_MODEL, hn, D_MODEL);

        // xz = hn @ in_proj_w^T : [512, 3072]
        {
            dim3 g((2 * D_INNER + 127) / 128, NTOK / 64);
            gemm_wmma_bf16<<<g, 256, 0, stream>>>(
                hn, in_w + (size_t)l * 2 * D_INNER * D_MODEL, xzb,
                NTOK, 2 * D_INNER, D_MODEL, D_MODEL, D_MODEL, 2 * D_INNER,
                nullptr, 0);
        }

        conv_silu_kernel<<<(NTOK * D_INNER + 255) / 256, 256, 0, stream>>>(
            xzb, mask, conv_w + (size_t)l * D_INNER * CONV_KW,
            conv_b + (size_t)l * D_INNER, xcb);

        // proj = xc @ x_proj_w^T : [512, 80]
        {
            dim3 g((XPROJ_N + 127) / 128, NTOK / 64);
            gemm_wmma_bf16<<<g, 256, 0, stream>>>(
                xcb, xp_w + (size_t)l * XPROJ_N * D_INNER, prj,
                NTOK, XPROJ_N, D_INNER, D_INNER, D_INNER, XPROJ_N,
                nullptr, 0);
        }

        // dt = softplus(proj[:, :48] @ dt_proj_w^T + dt_proj_b) : [512, 1536]
        {
            dim3 g((D_INNER + 127) / 128, NTOK / 64);
            gemm_wmma_bf16<<<g, 256, 0, stream>>>(
                prj, dtp_w + (size_t)l * D_INNER * DT_RANK, dtb,
                NTOK, D_INNER, DT_RANK, XPROJ_N, DT_RANK, D_INNER,
                dtp_b + (size_t)l * D_INNER, 2);
        }

        scan_kernel<<<(BATCH * D_INNER + 255) / 256, 256, 0, stream>>>(
            dtb, xcb, xzb, prj,
            A_log + (size_t)l * D_INNER * D_STATE, Dp + (size_t)l * D_INNER, yb);

        // h += y @ out_proj_w^T : residual epilogue
        {
            dim3 g((D_MODEL + 127) / 128, NTOK / 64);
            gemm_wmma_bf16<<<g, 256, 0, stream>>>(
                yb, out_w + (size_t)l * D_MODEL * D_INNER, h,
                NTOK, D_MODEL, D_INNER, D_INNER, D_INNER, D_MODEL,
                nullptr, 1);
        }
    }

    rmsnorm_kernel<<<NTOK, 256, 0, stream>>>(h, normf, hn, D_MODEL);

    // logits = hn @ embed^T : [512, 50280]
    {
        dim3 g((VOCAB + 127) / 128, NTOK / 64);
        gemm_wmma_bf16<<<g, 256, 0, stream>>>(
            hn, embedW, logits,
            NTOK, VOCAB, D_MODEL, D_MODEL, D_MODEL, VOCAB,
            nullptr, 0);
    }
}
